// MyLayer_59923383714208
// MI455X (gfx1250) — matrix operand order repaired and checked by reading, not compiled, not run on hardware
//
#include <hip/hip_runtime.h>
#include <hip/hip_bf16.h>

// ---------------------------------------------------------------------------
// Problem: out[b,o] = sum_i inputs[b,i] * mean[indices[i,o]] + bias[o]
//   inputs : (B, IN)  f32      B=2048, IN=4096
//   indices: (IN, OUT) i32     OUT=4096
//   mean   : (IN,) f32
//   bias   : (OUT,) f32
// Plan:
//   1a. decode+transpose: Wt (OUT x IN) bf16, Wt[o,i] = bf16(mean[indices[i,o]])
//   1b. convert inputs to bf16 (A, row-major M x K)
//   2.  wave32 WMMA bf16 GEMM with async global->LDS staging + fused bias.
// ---------------------------------------------------------------------------

typedef __attribute__((ext_vector_type(16))) __bf16 v16bf;
typedef __attribute__((ext_vector_type(8)))  float  v8f;

__device__ __forceinline__ unsigned short f32_to_bf16_rne(float f) {
    unsigned int u = __float_as_uint(f);
    unsigned int r = (u + 0x7fffu + ((u >> 16) & 1u)) >> 16;
    return (unsigned short)r;
}

// ---------------------------------------------------------------------------
// Pass 1a: gather-decode W and transpose to N-major bf16 via LDS tile.
// ---------------------------------------------------------------------------
#define TT 64

__global__ void __launch_bounds__(256)
decode_w_transpose_kernel(const int* __restrict__ indices,
                          const float* __restrict__ mean,
                          unsigned short* __restrict__ wt,
                          int IN, int OUT) {
    __shared__ unsigned short tile[TT][TT + 1];   // +1 pad kills bank conflicts
    const int i0 = blockIdx.y * TT;   // K (IN) base
    const int j0 = blockIdx.x * TT;   // N (OUT) base
    const int t  = threadIdx.x;
    const int f  = t * 16;
    const int r  = f >> 6;            // 0..63
    const int c  = f & 63;            // {0,16,32,48}

    {
        const int* src = indices + (size_t)(i0 + r) * OUT + j0 + c;
#pragma unroll
        for (int q = 0; q < 4; ++q) {
            int4 id = *(const int4*)(src + q * 4);
            tile[r][c + q * 4 + 0] = f32_to_bf16_rne(mean[id.x]);
            tile[r][c + q * 4 + 1] = f32_to_bf16_rne(mean[id.y]);
            tile[r][c + q * 4 + 2] = f32_to_bf16_rne(mean[id.z]);
            tile[r][c + q * 4 + 3] = f32_to_bf16_rne(mean[id.w]);
        }
    }
    __syncthreads();
    {
        unsigned short* dst = wt + (size_t)(j0 + r) * IN + i0 + c;
#pragma unroll
        for (int q = 0; q < 4; ++q) {
            ushort4 w;
            w.x = tile[c + q * 4 + 0][r];
            w.y = tile[c + q * 4 + 1][r];
            w.z = tile[c + q * 4 + 2][r];
            w.w = tile[c + q * 4 + 3][r];
            *(ushort4*)(dst + q * 4) = w;
        }
    }
}

// ---------------------------------------------------------------------------
// Pass 1b: A_bf16 = bf16(inputs)
// ---------------------------------------------------------------------------
__global__ void __launch_bounds__(256)
convert_a_kernel(const float* __restrict__ a,
                 unsigned short* __restrict__ abf,
                 int n) {
    int i = (blockIdx.x * 256 + threadIdx.x) * 4;
    if (i + 3 < n) {
        float4 v = *(const float4*)(a + i);
        ushort4 w;
        w.x = f32_to_bf16_rne(v.x);
        w.y = f32_to_bf16_rne(v.y);
        w.z = f32_to_bf16_rne(v.z);
        w.w = f32_to_bf16_rne(v.w);
        *(ushort4*)(abf + i) = w;
    }
}

// ---------------------------------------------------------------------------
// Pass 2: C = A (MxK) @ Wt^T (Wt is NxK) + bias, f32 out.
// Block tile 128x128, BK=32. 256 threads = 8 waves (wave32).
// Wave grid 4(M) x 2(N): each wave -> 32x64 patch = 2x4 WMMA 16x16 tiles.
// ---------------------------------------------------------------------------
#define BM 128
#define BN 128
#define BK 32
#define LDT 40   // padded LDS row stride (elements); 80B -> conflict-free b128

union FragBF {
    unsigned int u[8];
    v16bf v;
};

__global__ void __launch_bounds__(256)
wmma_gemm_bias_kernel(const unsigned short* __restrict__ A,    // (M,K) bf16
                      const unsigned short* __restrict__ Bt,   // (N,K) bf16
                      const float* __restrict__ bias,          // (N,)
                      float* __restrict__ C,                   // (M,N) f32
                      int M, int N, int K) {
    // single combined LDS block so byte offsets are valid VDST values
    __shared__ __attribute__((aligned(16))) unsigned short smem[(BM + BN) * LDT];
    unsigned short* sA = smem;             // [BM][LDT]
    unsigned short* sB = smem + BM * LDT;  // [BN][LDT]

    const int tid   = threadIdx.x;
    const int lane  = tid & 31;
    const int wave  = tid >> 5;
    const int waveM = wave >> 1;   // 0..3 -> 32-row strip
    const int waveN = wave & 1;    // 0..1 -> 64-col strip

    const int blockM = blockIdx.y * BM;
    const int blockN = blockIdx.x * BN;

    const int mrow  = lane & 15;
    const int khalf = lane >> 4;

    // per-thread staging assignment: 16 contiguous K-elements of one row
    const int f  = tid * 16;
    const int sr = f >> 5;    // row 0..127
    const int sc = f & 31;    // col {0,16}
    const unsigned int ldsA = (unsigned int)((sr * LDT + sc) * 2);
    const unsigned int ldsB = (unsigned int)(((BM + sr) * LDT + sc) * 2);

    v8f acc[2][4] = {};

    for (int k0 = 0; k0 < K; k0 += BK) {
        // ---- async stage A tile (BM x BK) ----
        // NOTE: smem is passed as an operand so the LDS object escapes and the
        // compiler must honor the "memory" clobber for the fragment reloads.
        {
            unsigned long long ga =
                (unsigned long long)(size_t)(A + (size_t)(blockM + sr) * K + k0 + sc);
            asm volatile(
                "global_load_async_to_lds_b128 %0, %1, off\n\t"
                "global_load_async_to_lds_b128 %0, %1, off offset:16"
                :: "v"(ldsA), "v"(ga), "v"(smem) : "memory");
        }
        // ---- async stage B tile (BN x BK), K-contiguous rows of Wt ----
        {
            unsigned long long ga =
                (unsigned long long)(size_t)(Bt + (size_t)(blockN + sr) * K + k0 + sc);
            asm volatile(
                "global_load_async_to_lds_b128 %0, %1, off\n\t"
                "global_load_async_to_lds_b128 %0, %1, off offset:16"
                :: "v"(ldsB), "v"(ga), "v"(smem) : "memory");
        }
        // prefetch next K tile into near caches (global_prefetch_b8)
        if (k0 + BK < K) {
            __builtin_prefetch(A + (size_t)(blockM + sr) * K + k0 + BK + sc, 0, 3);
            __builtin_prefetch(Bt + (size_t)(blockN + sr) * K + k0 + BK + sc, 0, 3);
        }
        asm volatile("s_wait_asynccnt 0" ::: "memory");
        __syncthreads();

        // ---- A fragments (ISA 16-bit A 16x32 layout), 2 x ds_load_b128 each ----
        FragBF af[2];
#pragma unroll
        for (int mt = 0; mt < 2; ++mt) {
            const unsigned short* base = sA + (waveM * 32 + mt * 16 + mrow) * LDT;
#pragma unroll
            for (int v = 0; v < 8; ++v) {
                int kk = ((v >> 2) << 4) + (khalf << 3) + ((v & 3) << 1);
                af[mt].u[v] = *(const unsigned int*)(base + kk);
            }
        }
        // ---- B fragments: lane holds one column (ISA 16-bit K layout, same as A) ----
        FragBF bfrag[4];
#pragma unroll
        for (int nt = 0; nt < 4; ++nt) {
            const unsigned short* base =
                sB + (waveN * 64 + nt * 16 + (lane & 15)) * LDT + (khalf << 3);
#pragma unroll
            for (int v = 0; v < 8; ++v) {
                bfrag[nt].u[v] = *(const unsigned int*)(base + ((v >> 2) << 4) + ((v & 3) << 1));
            }
        }

        // ---- 8 WMMAs: 32x64 patch, K advanced by 32 ----
#pragma unroll
        for (int mt = 0; mt < 2; ++mt) {
#pragma unroll
            for (int nt = 0; nt < 4; ++nt) {
                acc[mt][nt] = __builtin_amdgcn_wmma_f32_16x16x32_bf16(
                    false, af[mt].v, false, bfrag[nt].v,
                    (short)0, acc[mt][nt], false, false);
            }
        }
        __syncthreads();
    }

    // ---- epilogue: fused bias add, ISA 16x16 f32 C/D layout ----
#pragma unroll
    for (int mt = 0; mt < 2; ++mt) {
#pragma unroll
        for (int nt = 0; nt < 4; ++nt) {
            int row0 = blockM + waveM * 32 + mt * 16 + (khalf << 3);
            int col  = blockN + waveN * 64 + nt * 16 + (lane & 15);
            float bv = bias[col];
#pragma unroll
            for (int v = 0; v < 8; ++v) {
                C[(size_t)(row0 + v) * N + col] = acc[mt][nt][v] + bv;
            }
        }
    }
}

// ---------------------------------------------------------------------------
// Host-side launcher
// ---------------------------------------------------------------------------
extern "C" void kernel_launch(void* const* d_in, const int* in_sizes, int n_in,
                              void* d_out, int out_size, void* d_ws, size_t ws_size,
                              hipStream_t stream) {
    const float* inputs  = (const float*)d_in[0];  // (B, IN)
    const int*   indices = (const int*)  d_in[1];  // (IN, OUT)
    const float* mean    = (const float*)d_in[2];  // (IN,)
    const float* bias    = (const float*)d_in[3];  // (OUT,)
    float* out = (float*)d_out;

    const int IN  = in_sizes[2];              // 4096
    const int OUT = in_sizes[3];              // 4096
    const int B   = in_sizes[0] / IN;         // 2048

    // workspace layout: [A_bf16 : B*IN] [Wt_bf16 : OUT*IN]
    unsigned short* Abf = (unsigned short*)d_ws;
    unsigned short* Wt  = Abf + (size_t)B * IN;

    // Pass 1a: decode + transpose W
    {
        dim3 grid(OUT / TT, IN / TT);
        decode_w_transpose_kernel<<<grid, 256, 0, stream>>>(indices, mean, Wt, IN, OUT);
    }
    // Pass 1b: convert inputs to bf16
    {
        int n = B * IN;
        int blocks = (n / 4 + 255) / 256;
        convert_a_kernel<<<blocks, 256, 0, stream>>>(inputs, Abf, n);
    }
    // Pass 2: WMMA GEMM + bias
    {
        dim3 grid(OUT / BN, B / BM);
        wmma_gemm_bias_kernel<<<grid, 256, 0, stream>>>(Abf, Wt, bias, out, B, OUT, IN);
    }
}
